// CausalSelfAttention_46067819217140
// MI455X (gfx1250) — matrix operand order repaired and checked by reading, not compiled, not run on hardware
//
#include <hip/hip_runtime.h>
#include <math.h>

// Problem constants (match reference)
#define BATCH 4
#define SEQ   2048
#define CH    1024
#define NH    16
#define DHEAD 64
#define MROWS (BATCH*SEQ)   // 8192

typedef __bf16 bf16_t;
typedef __attribute__((ext_vector_type(16))) __bf16 v16bf;
typedef __attribute__((ext_vector_type(8)))  __bf16 v8bf;
typedef __attribute__((ext_vector_type(8)))  float  v8f;
typedef __attribute__((ext_vector_type(4)))  unsigned v4u;
typedef __attribute__((ext_vector_type(8)))  unsigned v8u;

// ---------------------------------------------------------------------------
// WMMA helper: D = A(16x32 bf16) x B(32x16 bf16) + C(16x16 f32)
// ---------------------------------------------------------------------------
__device__ __forceinline__ v8f wmma_bf16(v16bf a, v16bf b, v8f c) {
  return __builtin_amdgcn_wmma_f32_16x16x32_bf16(
      /*neg_a=*/false, a, /*neg_b=*/false, b,
      /*c_mod=*/(short)0, c, /*reuse_a=*/false, /*reuse_b=*/false);
}

// A-matrix (16x32, bf16, row-major source with leading dim ld).
// ISA layout: lane<16 -> row=lane, K 0..7 in v0..3, K 16..23 in v4..7
//             lane>=16 -> row=lane-16, K 8..15 and 24..31.
__device__ __forceinline__ v16bf load_A_tile(const bf16_t* base, int ld) {
  const int lane = threadIdx.x & 31;
  const int hi   = lane >> 4;
  const int row  = lane & 15;
  const bf16_t* p = base + row * ld + hi * 8;
  v8bf lo = *(const v8bf*)(p);        // K = hi*8 + 0..7
  v8bf hh = *(const v8bf*)(p + 16);   // K = 16 + hi*8 + 0..7
  v16bf a;
#pragma unroll
  for (int j = 0; j < 8; ++j) { a[j] = lo[j]; a[j + 8] = hh[j]; }
  return a;
}

// B-matrix (32x16) from an NT source: B[k][n] = W[n][k], W row-major, ld = K-stride.
// ISA layout: lanes 0..15 hold K=0..7,16..23 for col n=lane; lanes 16..31 hold K=8..15,24..31.
__device__ __forceinline__ v16bf load_B_tile_nt(const bf16_t* base, int ld) {
  const int lane = threadIdx.x & 31;
  const int col  = lane & 15;
  const int kb   = (lane >> 4) * 8;
  const bf16_t* p = base + col * ld + kb;
  v8bf lo = *(const v8bf*)(p);
  v8bf hh = *(const v8bf*)(p + 16);
  v16bf b;
#pragma unroll
  for (int j = 0; j < 8; ++j) { b[j] = lo[j]; b[j + 8] = hh[j]; }
  return b;
}

// ---------------------------------------------------------------------------
// TDM: 2D tile (rows x cols bf16 elements, row stride in elements) -> LDS.
// D# packing per CDNA5 ISA; groups 2/3 zero.
// ---------------------------------------------------------------------------
__device__ __forceinline__ void tdm_load_2d_bf16(const void* gptr, unsigned lds_addr,
                                                 unsigned rows, unsigned cols,
                                                 unsigned row_stride, unsigned tensor_rows) {
  const unsigned long long ga = (unsigned long long)gptr;
  v4u g0;
  g0[0] = 1u;                                            // count=1 (valid user D#)
  g0[1] = lds_addr;                                      // lds_addr (byte offset)
  g0[2] = (unsigned)(ga & 0xffffffffu);                  // global_addr[31:0]
  g0[3] = (unsigned)((ga >> 32) & 0x01ffffffu) | (2u << 30); // addr[56:32] | type=2
  v8u g1;
  g1[0] = 1u << 16;                                      // data_size=1 (2 bytes)
  g1[1] = (cols & 0xffffu) << 16;                        // tensor_dim0[15:0]
  g1[2] = ((cols >> 16) & 0xffffu) | ((tensor_rows & 0xffffu) << 16); // dim0 hi | dim1 lo
  g1[3] = ((tensor_rows >> 16) & 0xffffu) | ((cols & 0xffffu) << 16); // dim1 hi | tile_dim0
  g1[4] = rows & 0xffffu;                                // tile_dim1 (tile_dim2=0)
  g1[5] = row_stride;                                    // tensor_dim0_stride[31:0]
  g1[6] = 0u;                                            // stride0 hi | stride1 lo
  g1[7] = 0u;
  v4u gz = {0u, 0u, 0u, 0u};
  asm volatile("tensor_load_to_lds %0, %1, %2, %3\n\t"
               "s_wait_tensorcnt 0x0"
               :: "s"(g0), "s"(g1), "s"(gz), "s"(gz)
               : "memory");
}

// ---------------------------------------------------------------------------
// fp32 -> bf16 convert
// ---------------------------------------------------------------------------
__global__ void f32_to_bf16_kernel(const float* __restrict__ in,
                                   bf16_t* __restrict__ out, int n) {
  int i = blockIdx.x * blockDim.x + threadIdx.x;
  int stride = gridDim.x * blockDim.x;
  for (; i < n; i += stride) out[i] = (bf16_t)in[i];
}

// ---------------------------------------------------------------------------
// QKV projection: qkv[m,e] = sum_c x[m,c] * Win[e,c] + b[e]
// One 16x16 output tile per wave. Q pre-scaled by 1/sqrt(Dh), stored [B,H,N,Dh];
// K stored [B,H,N,Dh]; V stored TRANSPOSED [B,H,Dh,N] so attention B-tiles are
// contiguous 16-byte loads.
// ---------------------------------------------------------------------------
__global__ void qkv_gemm_kernel(const bf16_t* __restrict__ xb,
                                const bf16_t* __restrict__ wib,
                                const float* __restrict__ bias,
                                bf16_t* __restrict__ qb,
                                bf16_t* __restrict__ kb,
                                bf16_t* __restrict__ vtb) {
  const int wave = blockIdx.x * (blockDim.x >> 5) + (threadIdx.x >> 5);
  const int tiles_n = (3 * CH) / 16;          // 192
  const int tm = wave / tiles_n;
  const int tn = wave % tiles_n;
  if (tm >= MROWS / 16) return;

  v8f c = {};
  const bf16_t* arow = xb + (size_t)(tm * 16) * CH;
  const bf16_t* wrow = wib + (size_t)(tn * 16) * CH;
#pragma unroll 4
  for (int k = 0; k < CH; k += 32) {
    __builtin_prefetch(arow + k + 256, 0, 3);   // locality 3 -> WGP-scope prefetch
    __builtin_prefetch(wrow + k + 256, 0, 3);
    v16bf a = load_A_tile(arow + k, CH);
    v16bf b = load_B_tile_nt(wrow + k, CH);
    c = wmma_bf16(a, b, c);
  }

  const int lane = threadIdx.x & 31;
  const int hi = lane >> 4, col = lane & 15;
  const int e = tn * 16 + col;
  const float bv = bias[e];
  const int part = e / CH;                    // 0=Q 1=K 2=V (uniform per tile)
  const int cc = e % CH;
  const int h = cc / DHEAD, d = cc % DHEAD;
  const float scale = (part == 0) ? 0.125f : 1.0f;   // 1/sqrt(64) folded into Q
#pragma unroll
  for (int r = 0; r < 8; ++r) {
    const int m = tm * 16 + r + hi * 8;
    const int bb = m / SEQ, nn = m % SEQ;
    const bf16_t val = (bf16_t)((c[r] + bv) * scale);
    if (part == 0)
      qb[(((size_t)(bb * NH + h)) * SEQ + nn) * DHEAD + d] = val;
    else if (part == 1)
      kb[(((size_t)(bb * NH + h)) * SEQ + nn) * DHEAD + d] = val;
    else
      vtb[(((size_t)(bb * NH + h)) * DHEAD + d) * SEQ + nn] = val;
  }
}

// ---------------------------------------------------------------------------
// Attention: one block per (b, 16-query tile), 16 waves = 16 heads.
// Q tile staged to LDS by the Tensor Data Mover. Two-pass online softmax with
// WMMA for QK^T and P.V. Head-averaged probs accumulated in a shared 128KB
// fp32 tile via ds_add_f32 (no global atomics), then written out coalesced.
// ---------------------------------------------------------------------------
__global__ void attn_kernel(const bf16_t* __restrict__ qb,
                            const bf16_t* __restrict__ kb,
                            const bf16_t* __restrict__ vtb,
                            const unsigned char* __restrict__ kpm,  // bool [B,N]
                            bf16_t* __restrict__ attnb,             // [B,N,H,Dh] bf16
                            float* __restrict__ weights) {          // [B,N,N] f32
  extern __shared__ float wacc[];                         // [16][SEQ] fp32 accumulator
  __shared__ __align__(16) bf16_t qlds[NH][16 * DHEAD];   // per-wave Q tile
  __shared__ __align__(16) bf16_t pls[NH][16 * 32];       // per-wave P bounce

  const int tid = threadIdx.x;
  const int wv = tid >> 5;                   // wave id == head id
  const int h = wv;
  const int qt = blockIdx.x % (SEQ / 16);
  const int b  = blockIdx.x / (SEQ / 16);
  const int lane = tid & 31;
  const int hi = lane >> 4, col = lane & 15;
  const int q0 = qt * 16;
  const int nkt = qt + 1;                    // causal: key tiles 0..qt (uniform)

  // zero the shared weights accumulator
  for (int idx = tid; idx < 16 * SEQ; idx += blockDim.x) wacc[idx] = 0.0f;
  __syncthreads();

  const bf16_t* qbase = qb + ((size_t)(b * NH + h) * SEQ + q0) * DHEAD;
  const bf16_t* kbase = kb + (size_t)(b * NH + h) * SEQ * DHEAD;
  const bf16_t* vtbase = vtb + (size_t)(b * NH + h) * DHEAD * SEQ;
  const unsigned char* pm = kpm + (size_t)b * SEQ;

  // stage this head's 16x64 Q tile into LDS with the TDM
  tdm_load_2d_bf16(qbase, (unsigned)(unsigned long long)&qlds[wv][0],
                   /*rows=*/16, /*cols=*/DHEAD, /*row_stride=*/DHEAD,
                   /*tensor_rows=*/SEQ);

  // Q tile as two A matrices (d = 0..31, 32..63); Q already holds 1/sqrt(Dh)
  const v16bf qa0 = load_A_tile(&qlds[wv][0], DHEAD);
  const v16bf qa1 = load_A_tile(&qlds[wv][32], DHEAD);

  // ---- pass 1: per-lane online softmax over this lane's key column stream
  float mx[8], sm[8];
#pragma unroll
  for (int r = 0; r < 8; ++r) { mx[r] = -__builtin_inff(); sm[r] = 0.0f; }

  for (int kt = 0; kt < nkt; ++kt) {
    const int k0 = kt * 16;
    v8f s = {};
    s = wmma_bf16(qa0, load_B_tile_nt(kbase + (size_t)k0 * DHEAD + 0, DHEAD), s);
    s = wmma_bf16(qa1, load_B_tile_nt(kbase + (size_t)k0 * DHEAD + 32, DHEAD), s);
    const int key = k0 + col;
    const bool pad = pm[key] != 0;
#pragma unroll
    for (int r = 0; r < 8; ++r) {
      const int m = q0 + r + hi * 8;
      const float v = (key > m || pad) ? -__builtin_inff() : s[r];
      const float nm = fmaxf(mx[r], v);
      if (nm > -__builtin_inff()) {
        sm[r] = sm[r] * __expf(mx[r] - nm) + __expf(v - nm);
        mx[r] = nm;
      }
    }
  }
  // merge (max,sum) across the 16 lanes that share each row (xor stays in half)
#pragma unroll
  for (int off = 1; off < 16; off <<= 1) {
#pragma unroll
    for (int r = 0; r < 8; ++r) {
      const float om = __shfl_xor(mx[r], off, 32);
      const float os = __shfl_xor(sm[r], off, 32);
      const float nm = fmaxf(mx[r], om);
      if (nm > -__builtin_inff()) {
        sm[r] = sm[r] * __expf(mx[r] - nm) + os * __expf(om - nm);
        mx[r] = nm;
      }
    }
  }
  float rs[8];
#pragma unroll
  for (int r = 0; r < 8; ++r) rs[r] = (sm[r] > 0.0f) ? 1.0f / sm[r] : 0.0f;

  // ---- pass 2: recompute scores -> probs, accumulate weights (LDS) and P.V
  v8f o[4] = {v8f{}, v8f{}, v8f{}, v8f{}};
  for (int kt0 = 0; kt0 < nkt; kt0 += 2) {
#pragma unroll
    for (int t = 0; t < 2; ++t) {
      const int kt = kt0 + t;
      if (kt < nkt) {
        const int k0 = kt * 16;
        v8f s = {};
        s = wmma_bf16(qa0, load_B_tile_nt(kbase + (size_t)k0 * DHEAD + 0, DHEAD), s);
        s = wmma_bf16(qa1, load_B_tile_nt(kbase + (size_t)k0 * DHEAD + 32, DHEAD), s);
        const int key = k0 + col;
        const bool pad = pm[key] != 0;
#pragma unroll
        for (int r = 0; r < 8; ++r) {
          const int ml = r + hi * 8;
          const int m = q0 + ml;
          const float v = (key > m || pad) ? 0.0f : __expf(s[r] - mx[r]) * rs[r];
          pls[wv][ml * 32 + t * 16 + col] = (bf16_t)v;
          atomicAdd(&wacc[ml * SEQ + key], v * (1.0f / NH));   // ds_add_f32
        }
      } else {
#pragma unroll
        for (int r = 0; r < 8; ++r)
          pls[wv][(r + hi * 8) * 32 + t * 16 + col] = (bf16_t)0.0f;
      }
    }
    // P tile (A layout, K = 32 keys) x V^T tiles; per-wave LDS is in-order
    const v16bf pa = load_A_tile(&pls[wv][0], 32);
#pragma unroll
    for (int n0 = 0; n0 < DHEAD; n0 += 16) {
      const v16bf vt = load_B_tile_nt(vtbase + (size_t)n0 * SEQ + kt0 * 16, SEQ);
      o[n0 / 16] = wmma_bf16(pa, vt, o[n0 / 16]);
    }
  }

  // store attn tile to [B,N,H,Dh] (== [B,N,C]) in bf16 for the out-proj GEMM
#pragma unroll
  for (int n0 = 0; n0 < DHEAD; n0 += 16) {
#pragma unroll
    for (int r = 0; r < 8; ++r) {
      const int m = q0 + r + hi * 8;
      attnb[(((size_t)(b * SEQ + m)) * NH + h) * DHEAD + n0 + col] = (bf16_t)(o[n0 / 16][r]);
    }
  }

  // write head-averaged prob rows, coalesced
  __syncthreads();
  for (int idx = tid; idx < 16 * SEQ; idx += blockDim.x) {
    const int rq = idx >> 11;          // / SEQ
    const int kk = idx & (SEQ - 1);
    weights[((size_t)b * SEQ + q0 + rq) * SEQ + kk] = wacc[idx];
  }
}

// ---------------------------------------------------------------------------
// Output projection: out[m,o] = sum_c attn[m,c] * Wout[o,c] + b[o]  (fp32 out)
// ---------------------------------------------------------------------------
__global__ void out_gemm_kernel(const bf16_t* __restrict__ attnb,
                                const bf16_t* __restrict__ wob,
                                const float* __restrict__ bias,
                                float* __restrict__ out) {
  const int wave = blockIdx.x * (blockDim.x >> 5) + (threadIdx.x >> 5);
  const int tiles_n = CH / 16;                // 64
  const int tm = wave / tiles_n;
  const int tn = wave % tiles_n;
  if (tm >= MROWS / 16) return;

  v8f c = {};
  const bf16_t* arow = attnb + (size_t)(tm * 16) * CH;
  const bf16_t* wrow = wob + (size_t)(tn * 16) * CH;
#pragma unroll 4
  for (int k = 0; k < CH; k += 32) {
    __builtin_prefetch(arow + k + 256, 0, 3);   // locality 3 -> WGP-scope prefetch
    __builtin_prefetch(wrow + k + 256, 0, 3);
    v16bf a = load_A_tile(arow + k, CH);
    v16bf b = load_B_tile_nt(wrow + k, CH);
    c = wmma_bf16(a, b, c);
  }
  const int lane = threadIdx.x & 31;
  const int hi = lane >> 4, col = lane & 15;
  const float bv = bias[tn * 16 + col];
#pragma unroll
  for (int r = 0; r < 8; ++r) {
    const int m = tm * 16 + r + hi * 8;
    out[(size_t)m * CH + tn * 16 + col] = c[r] + bv;
  }
}

// ---------------------------------------------------------------------------
extern "C" void kernel_launch(void* const* d_in, const int* in_sizes, int n_in,
                              void* d_out, int out_size, void* d_ws, size_t ws_size,
                              hipStream_t stream) {
  (void)in_sizes; (void)n_in; (void)out_size; (void)ws_size;
  const float* x            = (const float*)d_in[0];
  const unsigned char* kpm  = (const unsigned char*)d_in[1];  // bool mask (1 byte/elem)
  const float* w_in         = (const float*)d_in[2];
  const float* b_in         = (const float*)d_in[3];
  const float* w_out        = (const float*)d_in[4];
  const float* b_out        = (const float*)d_in[5];

  float* out     = (float*)d_out;                       // [B,N,C]
  float* weights = out + (size_t)BATCH * SEQ * CH;      // [B,N,N]

  // workspace carve-out (~92 MB)
  char* p = (char*)d_ws;
  bf16_t* xb  = (bf16_t*)p; p += (size_t)MROWS * CH * sizeof(bf16_t);
  bf16_t* wib = (bf16_t*)p; p += (size_t)3 * CH * CH * sizeof(bf16_t);
  bf16_t* wob = (bf16_t*)p; p += (size_t)CH * CH * sizeof(bf16_t);
  bf16_t* qb  = (bf16_t*)p; p += (size_t)BATCH * NH * SEQ * DHEAD * sizeof(bf16_t);
  bf16_t* kb  = (bf16_t*)p; p += (size_t)BATCH * NH * SEQ * DHEAD * sizeof(bf16_t);
  bf16_t* vtb = (bf16_t*)p; p += (size_t)BATCH * NH * SEQ * DHEAD * sizeof(bf16_t);
  bf16_t* attnb = (bf16_t*)p;

  // 1) stage bf16 copies
  f32_to_bf16_kernel<<<4096, 256, 0, stream>>>(x, xb, MROWS * CH);
  f32_to_bf16_kernel<<<4096, 256, 0, stream>>>(w_in, wib, 3 * CH * CH);
  f32_to_bf16_kernel<<<2048, 256, 0, stream>>>(w_out, wob, CH * CH);

  // 2) QKV projection: 512 x 192 tiles, 4 waves/block
  qkv_gemm_kernel<<<(MROWS / 16) * ((3 * CH) / 16) / 4, 128, 0, stream>>>(
      xb, wib, b_in, qb, kb, vtb);

  // 3) attention: one block per (b, q-tile), 16 waves = 16 heads,
  //    128 KB dynamic LDS for the head-averaged prob accumulator
  attn_kernel<<<BATCH * (SEQ / 16), NH * 32, (size_t)16 * SEQ * sizeof(float), stream>>>(
      qb, kb, vtb, kpm, attnb, weights);

  // 4) output projection: 512 x 64 tiles, 4 waves/block
  out_gemm_kernel<<<(MROWS / 16) * (CH / 16) / 4, 128, 0, stream>>>(
      attnb, wob, b_out, out);
}
